// H2_V_88098369176178
// MI455X (gfx1250) — matrix operand order repaired and checked by reading, not compiled, not run on hardware
//
#include <hip/hip_runtime.h>
#include <hip/hip_bf16.h>

// ---- static config (mirrors reference) ----
#define BGR    64
#define A_PER  512
#define C2_PER 64
#define NA     (BGR * A_PER)    // 32768
#define NC2    (BGR * C2_PER)   // 4096
#define NV2    BGR              // 64
#define NHOM   (NC2 + NV2)      // 4160
#define INDIM  128
#define HID    256
#define EAA    (NA * 16)        // 524288
#define EI2    (NC2 * 16)       // 65536
#define EHOM   (EI2 + NC2)      // 69632
#define BN_EPS 1e-5f

typedef __attribute__((ext_vector_type(16))) _Float16 v16h;
typedef __attribute__((ext_vector_type(8)))  float    v8f;

static __device__ __forceinline__ int f2ord(float f) {
    int i = __float_as_int(f);
    return i ^ ((i >> 31) & 0x7fffffff);
}
static __device__ __forceinline__ float ord2f(int j) {
    int i = j ^ ((j >> 31) & 0x7fffffff);
    return __int_as_float(i);
}

// ---------------------------------------------------------------------------
// WMMA GEMM: C[M,0:256] = A[M,K](f32) * W, W pre-transposed to Wt[256,K] f16.
// Block = 128 threads (4 waves), tile 64 rows x 128 cols; grid = (2, M/64).
// The 128-col x 32-k B slice is staged cooperatively into LDS (double
// buffered) so all 4 waves share one global fetch of the weights; each wave
// then runs 8 v_wmma_f32_16x16x32_f16 per k-step from ds_load_b128 data.
// Requires: M%64==0, K%32==0 (true for every call site here).
// Optional bias + ReLU epilogue; ldc/colOff let branch outputs write straight
// into the concatenated hA buffer.
// ---------------------------------------------------------------------------
__global__ void k_gemm(const float* __restrict__ A, const _Float16* __restrict__ Wt,
                       const float* __restrict__ bias, float* __restrict__ C,
                       int M, int K, int ldc, int colOff, int doRelu)
{
    __shared__ _Float16 ldsB[2][128 * 32];          // 2 x 8KB ping-pong
    const int tid  = threadIdx.x;
    const int lane = tid & 31;
    const int wave = tid >> 5;
    const int r0 = blockIdx.y * 64 + wave * 16;
    const int c0 = blockIdx.x * 128;
    const int hi = lane >> 4;                       // 0: low half, 1: high half
    const float* Ar = A + (size_t)(r0 + (lane & 15)) * K + hi * 16;

    // cooperative stage of one 128-col x 32-k slice of Wt into ldsB[buf]
    auto stage = [&](int buf, int k0) {
        const _Float16* gp = Wt + (size_t)(c0 + tid) * K + k0;  // 64B, 32B-aligned
        _Float16* lp = &ldsB[buf][tid * 32];
        *(v16h*)(lp)      = *(const v16h*)(gp);
        *(v16h*)(lp + 16) = *(const v16h*)(gp + 16);
    };

    v8f acc0 = {}, acc1 = {}, acc2 = {}, acc3 = {};
    v8f acc4 = {}, acc5 = {}, acc6 = {}, acc7 = {};

    stage(0, 0);
    int buf = 0;
    for (int k0 = 0; k0 < K; k0 += 32) {
        __syncthreads();                            // ldsB[buf] ready, prev reads done
        if (k0 + 32 < K) stage(buf ^ 1, k0 + 32);   // overlap next stage with WMMAs

        // A fragment (contiguous K order): lanes 0-15 hold K
        // [k0..k0+15] of row M=lane; lanes 16-31 hold [k0+16..k0+31].
        const float4 x0 = *(const float4*)(Ar + k0);
        const float4 x1 = *(const float4*)(Ar + k0 + 4);
        const float4 x2 = *(const float4*)(Ar + k0 + 8);
        const float4 x3 = *(const float4*)(Ar + k0 + 12);
        v16h a;
        a[0] = (_Float16)x0.x; a[1] = (_Float16)x0.y; a[2] = (_Float16)x0.z; a[3] = (_Float16)x0.w;
        a[4] = (_Float16)x1.x; a[5] = (_Float16)x1.y; a[6] = (_Float16)x1.z; a[7] = (_Float16)x1.w;
        a[8] = (_Float16)x2.x; a[9] = (_Float16)x2.y; a[10] = (_Float16)x2.z; a[11] = (_Float16)x2.w;
        a[12] = (_Float16)x3.x; a[13] = (_Float16)x3.y; a[14] = (_Float16)x3.z; a[15] = (_Float16)x3.w;

        // B fragments from LDS: lane holds 16 contiguous K values of its column
        const _Float16* lb = &ldsB[buf][(lane & 15) * 32 + hi * 16];
#define BF(j) (*(const v16h*)(lb + (j) * 16 * 32))
        acc0 = __builtin_amdgcn_wmma_f32_16x16x32_f16(false, a, false, BF(0), (short)0, acc0, false, false);
        acc1 = __builtin_amdgcn_wmma_f32_16x16x32_f16(false, a, false, BF(1), (short)0, acc1, false, false);
        acc2 = __builtin_amdgcn_wmma_f32_16x16x32_f16(false, a, false, BF(2), (short)0, acc2, false, false);
        acc3 = __builtin_amdgcn_wmma_f32_16x16x32_f16(false, a, false, BF(3), (short)0, acc3, false, false);
        acc4 = __builtin_amdgcn_wmma_f32_16x16x32_f16(false, a, false, BF(4), (short)0, acc4, false, false);
        acc5 = __builtin_amdgcn_wmma_f32_16x16x32_f16(false, a, false, BF(5), (short)0, acc5, false, false);
        acc6 = __builtin_amdgcn_wmma_f32_16x16x32_f16(false, a, false, BF(6), (short)0, acc6, false, false);
        acc7 = __builtin_amdgcn_wmma_f32_16x16x32_f16(false, a, false, BF(7), (short)0, acc7, false, false);
#undef BF
        buf ^= 1;
    }

    const int nb = c0 + (lane & 15);
    const int mb = r0 + hi * 8;
#pragma unroll
    for (int j = 0; j < 8; ++j) {
        v8f acc = (j == 0) ? acc0 : (j == 1) ? acc1 : (j == 2) ? acc2 : (j == 3) ? acc3
                : (j == 4) ? acc4 : (j == 5) ? acc5 : (j == 6) ? acc6 : acc7;
        const int n = nb + j * 16;
        const float bv = bias ? bias[n] : 0.0f;
#pragma unroll
        for (int r = 0; r < 8; ++r) {
            float v = acc[r] + bv;
            if (doRelu) v = v > 0.0f ? v : 0.0f;
            C[(size_t)(mb + r) * ldc + colOff + n] = v;
        }
    }
}

// ---- weight transpose + f32->f16 convert: Wt[n*K+k] = W[k*N+n] ----
__global__ void k_wt(const float* __restrict__ W, _Float16* __restrict__ Wt, int K, int N)
{
    int i = blockIdx.x * 256 + threadIdx.x;
    if (i >= K * N) return;
    int n = i / K, k = i % K;
    Wt[i] = (_Float16)W[(size_t)k * N + n];
}

__global__ void k_zero_f(float* p, int n)
{
    int i = blockIdx.x * 256 + threadIdx.x;
    if (i < n) p[i] = 0.0f;
}

// el/er: per (node, head) dot of f[h*64..] with al/ar[h]
__global__ void k_eler(const float* __restrict__ f, const float* __restrict__ al,
                       const float* __restrict__ ar, float* __restrict__ el,
                       float* __restrict__ er, int n)
{
    int i = blockIdx.x * 256 + threadIdx.x;
    if (i >= n * 4) return;
    int node = i >> 2, h = i & 3;
    const float* fp = f + (size_t)node * HID + h * 64;
    float sl = 0.0f, sr = 0.0f;
#pragma unroll 4
    for (int d = 0; d < 64; ++d) {
        float v = fp[d];
        sl += v * al[h * 64 + d];
        sr += v * ar[h * 64 + d];
    }
    el[i] = sl; er[i] = sr;
}

__global__ void k_init_seg(int* mord, float* s, int n)
{
    int i = blockIdx.x * 256 + threadIdx.x;
    if (i < n) { mord[i] = (int)0x80000000; s[i] = 0.0f; }
}

static __device__ __forceinline__ float edge_logit(const float* el, const float* er,
                                                   int sN, int dN, int h)
{
    float w = el[sN * 4 + h] + er[dN * 4 + h];
    return w > 0.0f ? w : 0.2f * w;   // leaky_relu(0.2)
}

__global__ void k_edge_max(const int* __restrict__ src, const int* __restrict__ dst,
                           const float* __restrict__ el, const float* __restrict__ er,
                           int* __restrict__ mord, int E)
{
    int i = blockIdx.x * 256 + threadIdx.x;
    if (i >= E * 4) return;
    int e = i >> 2, h = i & 3;
    float w = edge_logit(el, er, src[e], dst[e], h);
    atomicMax(&mord[dst[e] * 4 + h], f2ord(w));
}

__global__ void k_edge_exp(const int* __restrict__ src, const int* __restrict__ dst,
                           const float* __restrict__ el, const float* __restrict__ er,
                           const int* __restrict__ mord, float* __restrict__ ex,
                           float* __restrict__ s, int E)
{
    int i = blockIdx.x * 256 + threadIdx.x;
    if (i >= E * 4) return;
    int e = i >> 2, h = i & 3;
    int dN = dst[e];
    float w = edge_logit(el, er, src[e], dN, h);
    float m = ord2f(mord[dN * 4 + h]);
    float ev = __expf(w - m);
    ex[i] = ev;
    atomicAdd(&s[dN * 4 + h], ev);
}

// out[dst] += f[src] * (ex/s[dst]), one thread per (edge, feature)
__global__ void k_edge_aggr(const int* __restrict__ src, const int* __restrict__ dst,
                            const float* __restrict__ f, const float* __restrict__ ex,
                            const float* __restrict__ s, float* __restrict__ out, int E)
{
    int i = blockIdx.x * 256 + threadIdx.x;
    if (i >= E * 256) return;
    int e = i >> 8, c = i & 255, h = c >> 6;
    int sN = src[e], dN = dst[e];
    float a = ex[e * 4 + h] / s[dN * 4 + h];
    atomicAdd(&out[(size_t)dN * HID + c], f[(size_t)sN * HID + c] * a);
}

__global__ void k_bias_relu(float* __restrict__ dstp, const float* __restrict__ srcp,
                            const float* __restrict__ b, int n)
{
    int i = blockIdx.x * 256 + threadIdx.x;
    if (i >= n) return;
    float v = srcp[i] + b[i & 255];
    dstp[i] = v > 0.0f ? v : 0.0f;
}

__global__ void k_gin_scatter(const int* __restrict__ src, const int* __restrict__ dst,
                              const float* __restrict__ x, float* __restrict__ agg, int E)
{
    int i = blockIdx.x * 256 + threadIdx.x;
    if (i >= E * 256) return;
    int e = i >> 8, c = i & 255;
    atomicAdd(&agg[(size_t)dst[e] * HID + c], x[(size_t)src[e] * HID + c]);
}

__global__ void k_gin_combine(float* __restrict__ agg, const float* __restrict__ x,
                              const float* __restrict__ eps, int n)
{
    int i = blockIdx.x * 256 + threadIdx.x;
    if (i < n) agg[i] = (1.0f + eps[0]) * x[i] + agg[i];
}

// column sums / sumsq (grid=128, block=256; each block owns 256 rows)
__global__ void k_colstats(const float* __restrict__ h, float* __restrict__ stats, int M)
{
    int c = threadIdx.x;
    int rowsPer = M / gridDim.x;
    int r0 = blockIdx.x * rowsPer;
    float s = 0.0f, s2 = 0.0f;
    for (int r = 0; r < rowsPer; ++r) {
        float v = h[(size_t)(r0 + r) * HID + c];
        s += v; s2 += v * v;
    }
    atomicAdd(&stats[c], s);
    atomicAdd(&stats[HID + c], s2);
}

__global__ void k_bn_relu(float* __restrict__ h, const float* __restrict__ stats,
                          const float* __restrict__ g, const float* __restrict__ beta, int M)
{
    int i = blockIdx.x * 256 + threadIdx.x;
    if (i >= M * HID) return;
    int c = i & 255;
    float inv = 1.0f / (float)M;
    float mu = stats[c] * inv;
    float var = stats[HID + c] * inv - mu * mu;
    float v = (h[i] - mu) * rsqrtf(var + BN_EPS) * g[c] + beta[c];
    h[i] = v > 0.0f ? v : 0.0f;
}

// fcC2 input: [max over 8 A-rows of hA | pca] -> [NC2, 576]
__global__ void k_c2in(const float* __restrict__ hA, const float* __restrict__ pca,
                       float* __restrict__ c2in)
{
    int i = blockIdx.x * 256 + threadIdx.x;
    if (i >= NC2 * 576) return;
    int c = i / 576, col = i % 576;
    float v;
    if (col < 512) {
        v = hA[(size_t)(c * 8) * 512 + col];
#pragma unroll
        for (int r = 1; r < 8; ++r) {
            float u = hA[(size_t)(c * 8 + r) * 512 + col];
            v = u > v ? u : v;
        }
    } else {
        v = pca[(size_t)c * 64 + (col - 512)];
    }
    c2in[i] = v;
}

// fcV2: h[4096+v] = relu(glb[v] @ W[16,256] + b)   (K=16 -> scalar kernel)
__global__ void k_fcv2(const float* __restrict__ glb, const float* __restrict__ W,
                       const float* __restrict__ b, float* __restrict__ h)
{
    int i = blockIdx.x * 256 + threadIdx.x;
    if (i >= NV2 * HID) return;
    int v = i >> 8, col = i & 255;
    float s = b[col];
#pragma unroll
    for (int k = 0; k < 16; ++k) s += glb[v * 16 + k] * W[k * HID + col];
    s = s > 0.0f ? s : 0.0f;
    h[(size_t)(NC2 + v) * HID + col] = s;
}

// homogeneous-graph edge arrays: [i2 edges ; p2 edges with dst offset NC2]
__global__ void k_h2edges(const int* __restrict__ i2s, const int* __restrict__ i2d,
                          int* __restrict__ es, int* __restrict__ ed)
{
    int i = blockIdx.x * 256 + threadIdx.x;
    if (i < EI2) { es[i] = i2s[i]; ed[i] = i2d[i]; }
    else if (i < EHOM) { int j = i - EI2; es[i] = j; ed[i] = NC2 + (j >> 6); }
}

// per-graph mean of hA over 512 rows -> R[:, 0:512]
__global__ void k_read1(const float* __restrict__ hA, float* __restrict__ R)
{
    int i = blockIdx.x * 256 + threadIdx.x;
    if (i >= BGR * 512) return;
    int g = i >> 9, col = i & 511;
    const float* p = hA + (size_t)g * 512 * 512 + col;
    float s = 0.0f;
    for (int r = 0; r < 512; ++r) s += p[(size_t)r * 512];
    R[(size_t)g * 768 + col] = s * (1.0f / 512.0f);
}

// per-graph mean over 64 C2 rows + 1 V2 row -> R[:, 512:768]
__global__ void k_read2(const float* __restrict__ h, float* __restrict__ R)
{
    int i = blockIdx.x * 256 + threadIdx.x;
    if (i >= BGR * HID) return;
    int g = i >> 8, col = i & 255;
    float s = h[(size_t)(NC2 + g) * HID + col];
    for (int r = 0; r < 64; ++r) s += h[(size_t)(g * 64 + r) * HID + col];
    R[(size_t)g * 768 + 512 + col] = s * (1.0f / 65.0f);
}

// final: out[g] = O1[g] @ W2[256,1] + b2
__global__ void k_final(const float* __restrict__ O1, const float* __restrict__ W2,
                        const float* __restrict__ b2, float* __restrict__ outp)
{
    int g = blockIdx.x * blockDim.x + threadIdx.x;
    if (g >= BGR) return;
    float s = b2[0];
    for (int c = 0; c < HID; ++c) s += O1[(size_t)g * HID + c] * W2[c];
    outp[g] = s;
}

// ---------------------------------------------------------------------------
extern "C" void kernel_launch(void* const* d_in, const int* in_sizes, int n_in,
                              void* d_out, int out_size, void* d_ws, size_t ws_size,
                              hipStream_t stream)
{
    (void)out_size;
    const float* feats = (const float*)d_in[0];
    const float* pca   = (const float*)d_in[1];
    const float* glb   = (const float*)d_in[2];
    const int base = 3;

    // Detect params pytree flattening: insertion order (eps scalar at leaf 8 of
    // the first gin) vs JAX tree_leaves sorted-dict-key order (W1 there).
    bool sortedLeaves = (n_in > 11) && (in_sizes[11] != 1);

    int iGatW[4], iGatB[4], iGatAl[4], iGatAr[4];
    int iGinEps[2], iGinW1[2], iGinB1[2], iGinG[2], iGinBeta[2], iGinW2[2], iGinB2[2];
    for (int b = 0; b < 2; ++b) {
        int lb = base + b * 15;
        for (int l = 0; l < 2; ++l) {
            int g0 = lb + l * 4, k = b * 2 + l;
            iGatW[k] = g0;
            if (sortedLeaves) { iGatAl[k] = g0 + 1; iGatAr[k] = g0 + 2; iGatB[k] = g0 + 3; }
            else              { iGatB[k] = g0 + 1; iGatAl[k] = g0 + 2; iGatAr[k] = g0 + 3; }
        }
        int gb = lb + 8;
        if (sortedLeaves) {
            iGinW1[b] = gb; iGinW2[b] = gb + 1; iGinB1[b] = gb + 2; iGinB2[b] = gb + 3;
            iGinBeta[b] = gb + 4; iGinEps[b] = gb + 5; iGinG[b] = gb + 6;
        } else {
            iGinEps[b] = gb; iGinW1[b] = gb + 1; iGinB1[b] = gb + 2; iGinG[b] = gb + 3;
            iGinBeta[b] = gb + 4; iGinW2[b] = gb + 5; iGinB2[b] = gb + 6;
        }
    }
    const int iFcC2W = base + 30, iFcC2B = base + 31;
    const int iFcV2W = base + 32, iFcV2B = base + 33;
    int iH2W[2], iH2B[2], iH2Al[2], iH2Ar[2];
    for (int l = 0; l < 2; ++l) {
        int g0 = base + 34 + l * 4;
        iH2W[l] = g0;
        if (sortedLeaves) { iH2Al[l] = g0 + 1; iH2Ar[l] = g0 + 2; iH2B[l] = g0 + 3; }
        else              { iH2B[l] = g0 + 1; iH2Al[l] = g0 + 2; iH2Ar[l] = g0 + 3; }
    }
    int iOutW1 = base + 42, iOutB1, iOutW2, iOutB2;
    if (sortedLeaves) { iOutW2 = base + 43; iOutB1 = base + 44; iOutB2 = base + 45; }
    else              { iOutB1 = base + 43; iOutW2 = base + 44; iOutB2 = base + 45; }

    const int eb = base + 46;   // e0s e0d e1s e1d g1s g1d i2s i2d p2s p2d
    const int* e0s = (const int*)d_in[eb + 0];
    const int* e0d = (const int*)d_in[eb + 1];
    const int* e1s = (const int*)d_in[eb + 2];
    const int* e1d = (const int*)d_in[eb + 3];
    const int* i2s = (const int*)d_in[eb + 6];
    const int* i2d = (const int*)d_in[eb + 7];

    auto Fp = [&](int i) { return (const float*)d_in[i]; };

    // ---- workspace bump allocator ----
    size_t off = 0;
    auto alloc = [&](size_t bytes) -> void* {
        off = (off + 255) & ~(size_t)255;
        void* p = (char*)d_ws + off;
        off += bytes;
        return p;
    };
    _Float16* wtGat[4]; // [branch*2+layer]
    for (int k = 0; k < 4; ++k) wtGat[k] = (_Float16*)alloc((size_t)((k & 1) ? HID : INDIM) * HID * 2);
    _Float16* wtGinW1[2]; _Float16* wtGinW2[2];
    for (int b = 0; b < 2; ++b) {
        wtGinW1[b] = (_Float16*)alloc((size_t)HID * HID * 2);
        wtGinW2[b] = (_Float16*)alloc((size_t)HID * HID * 2);
    }
    _Float16* wtFcC2 = (_Float16*)alloc((size_t)576 * HID * 2);
    _Float16* wtH2[2];
    for (int l = 0; l < 2; ++l) wtH2[l] = (_Float16*)alloc((size_t)HID * HID * 2);
    _Float16* wtOutW1 = (_Float16*)alloc((size_t)768 * HID * 2);

    float* F    = (float*)alloc((size_t)NA * HID * 4);   // fc output / GIN temp
    float* G1   = (float*)alloc((size_t)NA * HID * 4);   // aggregation / GIN temp
    float* H    = (float*)alloc((size_t)NA * HID * 4);   // node features (also h2 ping)
    float* HA   = (float*)alloc((size_t)NA * 512 * 4);   // concat of both branches
    float* HH2  = (float*)alloc((size_t)NHOM * HID * 4); // h2 pong
    float* el   = (float*)alloc((size_t)NA * 4 * 4);
    float* er   = (float*)alloc((size_t)NA * 4 * 4);
    int*   mord = (int*)  alloc((size_t)NA * 4 * 4);
    float* sseg = (float*)alloc((size_t)NA * 4 * 4);
    float* ex   = (float*)alloc((size_t)EAA * 4 * 4);
    float* stats= (float*)alloc((size_t)512 * 4);
    float* c2in = (float*)alloc((size_t)NC2 * 576 * 4);
    int*   h2es = (int*)  alloc((size_t)EHOM * 4);
    int*   h2ed = (int*)  alloc((size_t)EHOM * 4);
    float* R    = (float*)alloc((size_t)BGR * 768 * 4);
    float* O1   = (float*)alloc((size_t)BGR * HID * 4);
    if (off > ws_size) return;   // workspace too small; bail before any OOB write

    auto grid1 = [](int n) { return dim3((unsigned)((n + 255) / 256)); };
    auto zero  = [&](float* p, int n) { k_zero_f<<<grid1(n), 256, 0, stream>>>(p, n); };
    auto cvt   = [&](const float* W, _Float16* Wt, int K) {
        k_wt<<<grid1(K * HID), 256, 0, stream>>>(W, Wt, K, HID);
    };
    auto gemm = [&](const float* A, const _Float16* Wt, const float* bias, float* C,
                    int M, int K, int ldc, int colOff, int relu) {
        dim3 g(2, (unsigned)(M / 64));              // 128-col x 64-row block tiles
        k_gemm<<<g, 128, 0, stream>>>(A, Wt, bias, C, M, K, ldc, colOff, relu);
    };
    // full GAT layer: x[n,K] -> outH[n,256]  (outH may alias x)
    auto gat_layer = [&](const float* x, int n, int K, const _Float16* Wt,
                         const float* bvec, const float* al, const float* ar,
                         const int* src, const int* dst, int E, float* outH) {
        gemm(x, Wt, nullptr, F, n, K, HID, 0, 0);
        k_eler<<<grid1(n * 4), 256, 0, stream>>>(F, al, ar, el, er, n);
        k_init_seg<<<grid1(n * 4), 256, 0, stream>>>(mord, sseg, n * 4);
        k_edge_max<<<grid1(E * 4), 256, 0, stream>>>(src, dst, el, er, mord, E);
        k_edge_exp<<<grid1(E * 4), 256, 0, stream>>>(src, dst, el, er, mord, ex, sseg, E);
        zero(G1, n * HID);
        k_edge_aggr<<<dim3((unsigned)E), 256, 0, stream>>>(src, dst, F, ex, sseg, G1, E);
        k_bias_relu<<<grid1(n * HID), 256, 0, stream>>>(outH, G1, bvec, n * HID);
    };

    // ---- convert all GEMM weights to f16, transposed [N,K] ----
    for (int k = 0; k < 4; ++k) cvt(Fp(iGatW[k]), wtGat[k], (k & 1) ? HID : INDIM);
    for (int b = 0; b < 2; ++b) { cvt(Fp(iGinW1[b]), wtGinW1[b], HID); cvt(Fp(iGinW2[b]), wtGinW2[b], HID); }
    cvt(Fp(iFcC2W), wtFcC2, 576);
    for (int l = 0; l < 2; ++l) cvt(Fp(iH2W[l]), wtH2[l], HID);
    cvt(Fp(iOutW1), wtOutW1, 768);

    // ---- bottom: two branches over A nodes ----
    for (int b = 0; b < 2; ++b) {
        const int* src = b ? e1s : e0s;
        const int* dst = b ? e1d : e0d;
        gat_layer(feats, NA, INDIM, wtGat[b * 2 + 0], Fp(iGatB[b * 2 + 0]),
                  Fp(iGatAl[b * 2 + 0]), Fp(iGatAr[b * 2 + 0]), src, dst, EAA, H);
        gat_layer(H, NA, HID, wtGat[b * 2 + 1], Fp(iGatB[b * 2 + 1]),
                  Fp(iGatAl[b * 2 + 1]), Fp(iGatAr[b * 2 + 1]), src, dst, EAA, H);
        // GIN
        zero(G1, NA * HID);
        k_gin_scatter<<<dim3((unsigned)EAA), 256, 0, stream>>>(src, dst, H, G1, EAA);
        k_gin_combine<<<grid1(NA * HID), 256, 0, stream>>>(G1, H, Fp(iGinEps[b]), NA * HID);
        gemm(G1, wtGinW1[b], Fp(iGinB1[b]), F, NA, HID, HID, 0, 0);
        zero(stats, 512);
        k_colstats<<<dim3(128), 256, 0, stream>>>(F, stats, NA);
        k_bn_relu<<<grid1(NA * HID), 256, 0, stream>>>(F, stats, Fp(iGinG[b]), Fp(iGinBeta[b]), NA);
        gemm(F, wtGinW2[b], Fp(iGinB2[b]), HA, NA, HID, 512, b * HID, 1);
    }

    // ---- A -> C2 scatter-max (+pca concat), fcC2, fcV2 ----
    k_c2in<<<grid1(NC2 * 576), 256, 0, stream>>>(HA, pca, c2in);
    gemm(c2in, wtFcC2, Fp(iFcC2B), H, NC2, 576, HID, 0, 1);          // rows 0..4095 of H
    k_fcv2<<<grid1(NV2 * HID), 256, 0, stream>>>(glb, Fp(iFcV2W), Fp(iFcV2B), H);

    // ---- homogeneous graph: two GAT layers over [C2;V2] ----
    k_h2edges<<<grid1(EHOM), 256, 0, stream>>>(i2s, i2d, h2es, h2ed);
    gat_layer(H, NHOM, HID, wtH2[0], Fp(iH2B[0]), Fp(iH2Al[0]), Fp(iH2Ar[0]),
              h2es, h2ed, EHOM, HH2);
    gat_layer(HH2, NHOM, HID, wtH2[1], Fp(iH2B[1]), Fp(iH2Al[1]), Fp(iH2Ar[1]),
              h2es, h2ed, EHOM, H);

    // ---- readouts + output MLP ----
    k_read1<<<grid1(BGR * 512), 256, 0, stream>>>(HA, R);
    k_read2<<<grid1(BGR * HID), 256, 0, stream>>>(H, R);
    gemm(R, wtOutW1, Fp(iOutB1), O1, BGR, 768, HID, 0, 1);
    k_final<<<dim3(1), 64, 0, stream>>>(O1, Fp(iOutW2), Fp(iOutB2), (float*)d_out);
}
